// NonOverlappingLocallyConnected1d_24885040513346
// MI455X (gfx1250) — matrix operand order repaired and checked by reading, not compiled, not run on hardware
//
#include <hip/hip_runtime.h>

// Problem constants (from reference)
#define BS    64
#define CIN   128
#define COUT  128
#define OUTD  1024
#define Dd    4096          // OUT_DIM * BLK
#define NT    8             // n-values per workgroup (8*4 floats = 128B contiguous per row)
#define IC    8             // cin per pipeline stage -> K-chunk of 32 (one WMMA K step)
#define STAGES (CIN / IC)   // 16

// LDS geometry: row = 32 bf16 (kk = i_local*4+k), padded 64B -> 80B to spread
// the 16-lane fragment reads across banks.
#define ROWB   80
#define XROWS  (BS * NT)            // 512 rows
#define WROWS  (COUT * NT)          // 1024 rows
#define XBYTES (XROWS * ROWB)       // 40960
#define WBYTES (WROWS * ROWB)       // 81920
#define BUFBYTES (XBYTES + WBYTES)  // 122880; x2 buffers = 245760 <= 320KB WGP LDS

typedef __attribute__((ext_vector_type(4)))  __bf16 v4bf;
typedef __attribute__((ext_vector_type(4)))  float  fl4;
typedef __attribute__((ext_vector_type(8)))  __bf16 v8bf;
typedef __attribute__((ext_vector_type(16))) __bf16 v16bf;
typedef __attribute__((ext_vector_type(8)))  float  v8f;

union U16 { v8bf h[2]; v16bf v; };

// ---- stage global loads: each thread fetches 8 x-float4 + 16 w-float4 ------
__device__ __forceinline__ void stage_load(const float4* __restrict__ x4,
                                           const float4* __restrict__ w4,
                                           int s, int n0, int tid,
                                           float4* px, float4* pw) {
  const int i0 = s * IC;
#pragma unroll
  for (int it = 0; it < 8; ++it) {            // x: 64b * 8i rows, 8 float4/row
    int idx = it * 512 + tid;
    int j   = idx & 7;                        // n within slab (float4 = 4 k's of one n)
    int row = idx >> 3;
    int il  = row & 7;                        // i_local
    int b   = row >> 3;
    long f  = ((long)(b * CIN + i0 + il)) * Dd + (long)(n0 + j) * 4;
    px[it]  = x4[f >> 2];
  }
#pragma unroll
  for (int it = 0; it < 16; ++it) {           // w: 128o * 8i rows, 8 float4/row
    int idx = it * 512 + tid;
    int j   = idx & 7;
    int row = idx >> 3;
    int il  = row & 7;
    int o   = row >> 3;
    long f  = ((long)(o * CIN + i0 + il)) * Dd + (long)(n0 + j) * 4;
    pw[it]  = w4[f >> 2];
  }
}

__device__ __forceinline__ v4bf cvt4(float4 v) {
  fl4 s;
  s[0] = v.x; s[1] = v.y; s[2] = v.z; s[3] = v.w;
  return __builtin_convertvector(s, v4bf);   // packed v_cvt_pk_bf16_f32 path
}

// ---- convert f32 -> bf16 and stage into LDS (kk = i_local*4+k contiguous) --
__device__ __forceinline__ void stage_store(char* xb, char* wb, int tid,
                                            const float4* px, const float4* pw) {
#pragma unroll
  for (int it = 0; it < 8; ++it) {
    int idx = it * 512 + tid;
    int j = idx & 7, row = idx >> 3, il = row & 7, b = row >> 3;
    *(v4bf*)(xb + (b * NT + j) * ROWB + il * 8) = cvt4(px[it]);
  }
#pragma unroll
  for (int it = 0; it < 16; ++it) {
    int idx = it * 512 + tid;
    int j = idx & 7, row = idx >> 3, il = row & 7, o = row >> 3;
    *(v4bf*)(wb + (o * NT + j) * ROWB + il * 8) = cvt4(pw[it]);
  }
}

// ---- fragment loaders (ISA 7.12.2 16-bit A/B layouts) ----------------------
__device__ __forceinline__ void loadB(const char* wb, int ocol, int lsel, int n,
                                      U16* Bf) {
  // B (32x16 K x N): lanes 0-15: K=0..7,16..23 ; lanes 16-31: K=8..15,24..31
  const char* bbase = wb + (ocol * NT + n) * ROWB + lsel * 16;
  Bf->h[0] = *(const v8bf*)(bbase);
  Bf->h[1] = *(const v8bf*)(bbase + 32);
}

__device__ __forceinline__ void loadA(const char* xb, int brow, int lsel, int n,
                                      U16* Af) {
  // A (16x32 M x K): lanes 0-15: K=0..7,16..23 ; lanes 16-31: K=8..15,24..31
  const char* abase = xb + (brow * NT + n) * ROWB;
  Af->h[0] = *(const v8bf*)(abase + lsel * 16);
  Af->h[1] = *(const v8bf*)(abase + 32 + lsel * 16);
}

// ---- one K=32 step for all (2 b-tiles) x (8 n) tiles of this wave ----------
// Fragment loads are double-buffered so the ds_loads for step n+1 are in
// flight while the two WMMAs of step n execute (avoids s_wait_dscnt 0).
__device__ __forceinline__ void compute_stage(const char* xb, const char* wb,
                                              int lsel, int lr,
                                              int o_tile, int b_pair,
                                              v8f acc[2][NT]) {
  const int ocol = o_tile * 16 + lr;
  const int br0  = (b_pair * 2 + 0) * 16 + lr;
  const int br1  = (b_pair * 2 + 1) * 16 + lr;

  U16 B[2], A0[2], A1[2];
  loadB(wb, ocol, lsel, 0, &B[0]);
  loadA(xb, br0, lsel, 0, &A0[0]);
  loadA(xb, br1, lsel, 0, &A1[0]);
#pragma unroll
  for (int n = 0; n < NT; ++n) {
    const int cur = n & 1, nxt = cur ^ 1;
    if (n + 1 < NT) {                         // prefetch next step's fragments
      loadB(wb, ocol, lsel, n + 1, &B[nxt]);
      loadA(xb, br0, lsel, n + 1, &A0[nxt]);
      loadA(xb, br1, lsel, n + 1, &A1[nxt]);
    }
    acc[0][n] = __builtin_amdgcn_wmma_f32_16x16x32_bf16(
        false, A0[cur].v, false, B[cur].v, (short)0, acc[0][n], false, false);
    acc[1][n] = __builtin_amdgcn_wmma_f32_16x16x32_bf16(
        false, A1[cur].v, false, B[cur].v, (short)0, acc[1][n], false, false);
  }
}

__global__ __launch_bounds__(512, 1) void lc1d_wmma_kernel(
    const float* __restrict__ xg, const float* __restrict__ wg,
    float* __restrict__ outg) {
  extern __shared__ char smem[];
  const int tid    = threadIdx.x;
  const int lane   = tid & 31;
  const int wv     = tid >> 5;       // 16 waves
  const int o_tile = wv & 7;         // 8 o-tiles of 16
  const int b_pair = wv >> 3;        // covers b-tiles {2p, 2p+1}
  const int n0     = blockIdx.x * NT;
  const int lsel   = (lane < 16) ? 0 : 1;
  const int lr     = lane & 15;

  const float4* x4 = (const float4*)xg;
  const float4* w4 = (const float4*)wg;

  v8f acc[2][NT];
  const v8f vzero = {0.f, 0.f, 0.f, 0.f, 0.f, 0.f, 0.f, 0.f};
#pragma unroll
  for (int t = 0; t < 2; ++t)
#pragma unroll
    for (int n = 0; n < NT; ++n) acc[t][n] = vzero;

  float4 px[8], pw[16];

  // prologue: fill buffer 0
  stage_load(x4, w4, 0, n0, tid, px, pw);
  stage_store(smem, smem + XBYTES, tid, px, pw);
  __syncthreads();

  // double-buffered pipeline over cin
  for (int s = 0; s < STAGES; ++s) {
    const char* xb = smem + (s & 1) * BUFBYTES;
    const char* wb = xb + XBYTES;
    if (s + 1 < STAGES) stage_load(x4, w4, s + 1, n0, tid, px, pw);  // prefetch
    compute_stage(xb, wb, lsel, lr, o_tile, b_pair, acc);            // WMMAs
    if (s + 1 < STAGES) {
      char* xb2 = smem + ((s + 1) & 1) * BUFBYTES;
      stage_store(xb2, xb2 + XBYTES, tid, px, pw);
    }
    __syncthreads();
  }

  // epilogue: scale by 1/sqrt(cin); assemble 8 n's per (b,o) into 32B stores
  const float scale = 0.08838834764831845f;   // 1/sqrt(128)
  const int obase = o_tile * 16 + lr;
#pragma unroll
  for (int t = 0; t < 2; ++t) {
#pragma unroll
    for (int v = 0; v < 8; ++v) {
      // C/D layout: VGPR v -> row M=v (lanes 0-15) / M=v+8 (lanes 16-31)
      int b = b_pair * 32 + t * 16 + lsel * 8 + v;
      float4 f0, f1;
      f0.x = acc[t][0][v] * scale; f0.y = acc[t][1][v] * scale;
      f0.z = acc[t][2][v] * scale; f0.w = acc[t][3][v] * scale;
      f1.x = acc[t][4][v] * scale; f1.y = acc[t][5][v] * scale;
      f1.z = acc[t][6][v] * scale; f1.w = acc[t][7][v] * scale;
      float4* op = (float4*)(outg + ((long)(b * COUT + obase)) * OUTD + n0);
      op[0] = f0;
      op[1] = f1;
    }
  }
}

extern "C" void kernel_launch(void* const* d_in, const int* in_sizes, int n_in,
                              void* d_out, int out_size, void* d_ws, size_t ws_size,
                              hipStream_t stream) {
  (void)in_sizes; (void)n_in; (void)out_size; (void)d_ws; (void)ws_size;
  const float* x = (const float*)d_in[0];
  const float* w = (const float*)d_in[1];
  float* out = (float*)d_out;

  const size_t shmem = 2 * (size_t)BUFBYTES;  // 240 KB dynamic LDS (> 64KB static cap)
  hipFuncSetAttribute((const void*)lc1d_wmma_kernel,
                      hipFuncAttributeMaxDynamicSharedMemorySize, (int)shmem);

  dim3 grid(OUTD / NT);   // 128 workgroups, one per 8-n slab
  dim3 block(512);        // 16 wave32
  hipLaunchKernelGGL(lc1d_wmma_kernel, grid, block, shmem, stream, x, w, out);
}
